// LACF_33028298506953
// MI455X (gfx1250) — matrix operand order repaired and checked by reading, not compiled, not run on hardware
//
#include <hip/hip_runtime.h>
#include <hip/hip_bf16.h>

// ---------------------------------------------------------------------------
// Types for CDNA5 WMMA (wave32): v_wmma_f32_16x16x32_bf16
// ---------------------------------------------------------------------------
typedef __attribute__((ext_vector_type(16))) __bf16          v16bf;
typedef __attribute__((ext_vector_type(16))) unsigned short  v16us;   // 32 bytes
typedef __attribute__((ext_vector_type(8)))  float           v8f;

__device__ __forceinline__ unsigned short f32_bf16(float f) {
    unsigned int u = __builtin_bit_cast(unsigned int, f);
    unsigned int r = u + 0x7FFFu + ((u >> 16) & 1u);   // round-to-nearest-even
    return (unsigned short)(r >> 16);
}

__device__ __forceinline__ float sigmoidf_fast(float x) {
    return 1.0f / (1.0f + __expf(-x));
}

// A fragment: 16x32 bf16 from row-major LDS tile (lda elements).
// ISA 7.12.2: half h=lane>>4, row M=lane&15, K = (j>>2)*16 + h*8 + (j&3)*2 + s
// Per lane this is two contiguous 8-element (16B) runs -> 2x ds_load_b128.
__device__ __forceinline__ v16bf load_a_frag(const unsigned short* a, int lda, int kBase) {
    int lane = threadIdx.x & 31;
    int h = lane >> 4, m = lane & 15;
    v16us v;
#pragma unroll
    for (int j = 0; j < 8; ++j) {
        int k0 = kBase + ((j >> 2) << 4) + (h << 3) + ((j & 3) << 1);
        v[2 * j]     = a[m * lda + k0];
        v[2 * j + 1] = a[m * lda + k0 + 1];
    }
    return __builtin_bit_cast(v16bf, v);
}

// B fragment from pre-packed global weights: fragment-major, 32B per lane,
// fully coalesced (two global_load_b128 per lane, L2-resident).
__device__ __forceinline__ v16bf load_b_packed(const unsigned short* packed,
                                               int numK, int ks, int nt) {
    int lane = threadIdx.x & 31;
    const v16us* p = (const v16us*)packed + ((nt * numK + ks) * 32 + lane);
    return __builtin_bit_cast(v16bf, *p);
}

#define WMMA_BF16(a, b, c) \
    __builtin_amdgcn_wmma_f32_16x16x32_bf16(false, (a), false, (b), (short)0, (c), false, false)

// ---------------------------------------------------------------------------
// Weight packer: row-major f32 W[K][N] -> fragment-major bf16.
// frag = nt*(K/32)+ks ; lane element (j,s): K = ks*32 + (j>>2)*16 + (lane>>4)*8 + (j&3)*2+s,
// N = nt*16 + (lane&15). One 32B store per task, coalesced.
// ---------------------------------------------------------------------------
__global__ void pack_b_kernel(const float* __restrict__ W,
                              unsigned short* __restrict__ packed, int K, int N) {
    int tid = blockIdx.x * blockDim.x + threadIdx.x;
    int numK = K >> 5;
    int total = numK * (N >> 4) * 32;
    if (tid >= total) return;
    int lane = tid & 31;
    int frag = tid >> 5;
    int ks = frag % numK, nt = frag / numK;
    int h = lane >> 4, n = nt * 16 + (lane & 15);
    v16us v;
#pragma unroll
    for (int j = 0; j < 8; ++j) {
        int k = ks * 32 + ((j >> 2) << 4) + h * 8 + ((j & 3) << 1);
        v[2 * j]     = f32_bf16(W[k * N + n]);
        v[2 * j + 1] = f32_bf16(W[(k + 1) * N + n]);
    }
    *((v16us*)packed + frag * 32 + lane) = v;
}

// ---------------------------------------------------------------------------
// Trivial elementwise kernels
// ---------------------------------------------------------------------------
__global__ void zero_kernel(float* __restrict__ p, long n) {
    long i = blockIdx.x * (long)blockDim.x + threadIdx.x;
    long stride = (long)gridDim.x * blockDim.x;
    for (; i < n; i += stride) p[i] = 0.0f;
}

__global__ void init_kernel(const float* __restrict__ emb0, float* __restrict__ e,
                            float* __restrict__ s, long nd) {
    long i = blockIdx.x * (long)blockDim.x + threadIdx.x;
    long stride = (long)gridDim.x * blockDim.x;
    for (; i < nd; i += stride) {
        float v = emb0[i];
        e[i] = v; e[nd + i] = v; e[2 * nd + i] = v;
        s[i] = v; s[nd + i] = v; s[2 * nd + i] = v;
    }
}

__global__ void update_kernel(float* __restrict__ e, const float* __restrict__ gnn,
                              float* __restrict__ s, long nd) {
    long i = blockIdx.x * (long)blockDim.x + threadIdx.x;
    long stride = (long)gridDim.x * blockDim.x;
    for (; i < nd; i += stride) {
        float en = e[i] + gnn[i];
        e[i] = en;
        s[i] += en;
    }
}

// ---------------------------------------------------------------------------
// SpMM: out[h] += val * x[t]   (16 lanes per edge, float4 per lane)
// mode 0: val = vals[e] ; mode 1: val = rowsum[h] > 0 ? w[e]/rowsum[h] : 0
// ---------------------------------------------------------------------------
__global__ __launch_bounds__(256) void spmm_kernel(
    const float* __restrict__ vals, const float* __restrict__ wv,
    const float* __restrict__ rowsum,
    const int* __restrict__ h_idx, const int* __restrict__ t_idx,
    const float* __restrict__ x, float* __restrict__ out, long E, int mode) {
    long gid = blockIdx.x * (long)blockDim.x + threadIdx.x;
    long e = gid >> 4;
    if (e >= E) return;
    int c = (int)(gid & 15) * 4;
    int h = h_idx[e], t = t_idx[e];
    float v;
    if (mode == 0) {
        v = vals[e];
    } else {
        float rs = rowsum[h];
        v = (rs > 0.0f) ? (wv[e] / rs) : 0.0f;
    }
    const float4 xv = *(const float4*)(x + (long)t * 64 + c);
    float* o = out + (long)h * 64 + c;
    atomicAdd(o + 0, v * xv.x);
    atomicAdd(o + 1, v * xv.y);
    atomicAdd(o + 2, v * xv.z);
    atomicAdd(o + 3, v * xv.w);
}

// ---------------------------------------------------------------------------
// Edge MLP (branch 1): logit = relu([e1[h]|e1[t]] @ W1 + b1) @ W2 + b2
//   w = sigmoid(gumbel + logit);  rowsum[h] += w
// 256 thr = 8 waves; wave owns 16 edges. K=128 -> 4 bf16 WMMAs per N-tile.
// A-fragments hoisted (loaded once, reused across all 4 N-tiles).
// B-fragments come pre-packed from global (L2-resident, coalesced b128).
// Layer-2 [16,64]@[64,1] folded into the epilogue + shfl_xor tree.
// ---------------------------------------------------------------------------
__global__ __launch_bounds__(256) void edge_mlp_kernel(
    const float* __restrict__ e1, const int* __restrict__ h_idx,
    const int* __restrict__ t_idx,
    const unsigned short* __restrict__ pW1,   // packed [4 ks][4 nt]
    const float* __restrict__ b1, const float* __restrict__ W2,
    const float* __restrict__ b2, const float* __restrict__ gum,
    float* __restrict__ w_out, float* __restrict__ rowsum, int E) {
    __shared__ unsigned short sA[8][16 * 128];  // per-wave 16x128 bf16 tile
    __shared__ float sb1[64], sW2[64];

    const int tid = threadIdx.x;
    if (tid < 64) { sb1[tid] = b1[tid]; sW2[tid] = W2[tid]; }

    const int wave = tid >> 5, lane = tid & 31;
    const int eBase = blockIdx.x * 128 + wave * 16;
    unsigned short* A = sA[wave];

    // Gather 16x128 concat-feature tile (coalesced 128B rows from L2-resident e1)
#pragma unroll 4
    for (int j = 0; j < 64; ++j) {
        int lin = j * 32 + lane;
        int r = lin >> 7, c = lin & 127;
        int e = eBase + r; if (e >= E) e = E - 1;       // clamp: keep EXEC full
        int node = (c < 64) ? h_idx[e] : t_idx[e];
        A[r * 128 + c] = f32_bf16(e1[(long)node * 64 + (c & 63)]);
    }
    __syncthreads();

    v16bf afr[4];
#pragma unroll
    for (int ks = 0; ks < 4; ++ks) afr[ks] = load_a_frag(A, 128, ks * 32);

    float logitPart[8];
#pragma unroll
    for (int r = 0; r < 8; ++r) logitPart[r] = 0.0f;

#pragma unroll
    for (int nt = 0; nt < 4; ++nt) {
        v8f acc = {};
#pragma unroll
        for (int ks = 0; ks < 4; ++ks)
            acc = WMMA_BF16(afr[ks], load_b_packed(pW1, 4, ks, nt), acc);
        int n = (lane & 15) + nt * 16;
        float b1n = sb1[n], w2n = sW2[n];
#pragma unroll
        for (int r = 0; r < 8; ++r) {
            float v = acc[r] + b1n;
            v = v > 0.0f ? v : 0.0f;    // relu
            logitPart[r] += v * w2n;    // fold in layer-2 dot
        }
    }
    // reduce over the 16 N-lanes within each half-wave
#pragma unroll
    for (int m = 1; m < 16; m <<= 1)
#pragma unroll
        for (int r = 0; r < 8; ++r)
            logitPart[r] += __shfl_xor(logitPart[r], m, 32);

    if ((lane & 15) == 0) {
        int h = lane >> 4;
        float bb2 = b2[0];
#pragma unroll
        for (int r = 0; r < 8; ++r) {
            int e = eBase + r + h * 8;   // C layout: M = r + 8*half
            if (e < E) {
                float ww = sigmoidf_fast(gum[e] + logitPart[r] + bb2);
                w_out[e] = ww;
                atomicAdd(&rowsum[h_idx[e]], ww);
            }
        }
    }
}

// ---------------------------------------------------------------------------
// Node gate MLP (branch 2): gated = sigmoid(gumbel + relu(e2@W1+b1)@W2+b2) * e2
// 256 thr = 8 waves; wave owns 16 nodes; two chained bf16 WMMA GEMMs.
// ---------------------------------------------------------------------------
__global__ __launch_bounds__(256) void gate_kernel(
    const float* __restrict__ e2,
    const unsigned short* __restrict__ pW1,   // packed [2 ks][4 nt]
    const float* __restrict__ b1,
    const unsigned short* __restrict__ pW2,   // packed [2 ks][4 nt]
    const float* __restrict__ b2,
    const float* __restrict__ gum, float* __restrict__ gated, int N) {
    __shared__ unsigned short sA[8][16 * 64];
    __shared__ unsigned short sH[8][16 * 64];
    __shared__ float sb1[64], sb2[64];

    const int tid = threadIdx.x;
    if (tid < 64) { sb1[tid] = b1[tid]; sb2[tid] = b2[tid]; }

    const int wave = tid >> 5, lane = tid & 31;
    const int rowBase = blockIdx.x * 128 + wave * 16;
    unsigned short* A = sA[wave];
    unsigned short* H = sH[wave];

#pragma unroll 4
    for (int j = 0; j < 32; ++j) {
        int lin = j * 32 + lane;
        int r = lin >> 6, c = lin & 63;
        int node = rowBase + r; if (node >= N) node = N - 1;
        A[r * 64 + c] = f32_bf16(e2[(long)node * 64 + c]);
    }
    __syncthreads();

    // GEMM 1: H = relu(A @ W1 + b1)
    {
        v16bf afr[2];
#pragma unroll
        for (int ks = 0; ks < 2; ++ks) afr[ks] = load_a_frag(A, 64, ks * 32);
#pragma unroll
        for (int nt = 0; nt < 4; ++nt) {
            v8f acc = {};
#pragma unroll
            for (int ks = 0; ks < 2; ++ks)
                acc = WMMA_BF16(afr[ks], load_b_packed(pW1, 2, ks, nt), acc);
            int n = (lane & 15) + nt * 16;
            int hm = lane >> 4;
#pragma unroll
            for (int r = 0; r < 8; ++r) {
                float v = acc[r] + sb1[n];
                v = v > 0.0f ? v : 0.0f;
                H[(r + 8 * hm) * 64 + n] = f32_bf16(v);
            }
        }
    }
    __syncthreads();

    // GEMM 2: lg = H @ W2 + b2 ; gate & scale
    {
        v16bf hfr[2];
#pragma unroll
        for (int ks = 0; ks < 2; ++ks) hfr[ks] = load_a_frag(H, 64, ks * 32);
#pragma unroll
        for (int nt = 0; nt < 4; ++nt) {
            v8f acc = {};
#pragma unroll
            for (int ks = 0; ks < 2; ++ks)
                acc = WMMA_BF16(hfr[ks], load_b_packed(pW2, 2, ks, nt), acc);
            int n = (lane & 15) + nt * 16;
            int hm = lane >> 4;
#pragma unroll
            for (int r = 0; r < 8; ++r) {
                int node = rowBase + r + 8 * hm;
                if (node < N) {
                    float lg = acc[r] + sb2[n];
                    float g = sigmoidf_fast(gum[(long)node * 64 + n] + lg);
                    gated[(long)node * 64 + n] = g * e2[(long)node * 64 + n];
                }
            }
        }
    }
}

// ---------------------------------------------------------------------------
// Host driver
// ---------------------------------------------------------------------------
extern "C" void kernel_launch(void* const* d_in, const int* in_sizes, int n_in,
                              void* d_out, int out_size, void* d_ws, size_t ws_size,
                              hipStream_t stream) {
    const float* emb0        = (const float*)d_in[0];
    const int*   h_idx       = (const int*)d_in[1];
    const int*   t_idx       = (const int*)d_in[2];
    const float* G_values    = (const float*)d_in[3];
    const float* edge_gumbel = (const float*)d_in[4];
    const float* emb_gumbel  = (const float*)d_in[5];
    const float* edge_W1     = (const float*)d_in[6];
    const float* edge_b1     = (const float*)d_in[7];
    const float* edge_W2     = (const float*)d_in[8];
    const float* edge_b2     = (const float*)d_in[9];
    const float* emb_W1      = (const float*)d_in[10];
    const float* emb_b1      = (const float*)d_in[11];
    const float* emb_W2      = (const float*)d_in[12];
    const float* emb_b2      = (const float*)d_in[13];

    const long ND = in_sizes[0];           // N * 64
    const long E  = in_sizes[1];
    const int  N  = (int)(ND / 64);
    const int  L  = (int)(in_sizes[4] / E);

    float* ws     = (float*)d_ws;
    float* e      = ws;                    // 3 * ND  (e0,e1,e2)
    float* gnn    = ws + 3 * ND;           // ND      (reused per branch)
    float* gated  = ws + 4 * ND;           // ND
    float* wbuf   = ws + 5 * ND;           // E
    float* rowsum = ws + 5 * ND + E;       // N
    // packed bf16 weight regions (32B-aligned)
    uintptr_t pb = (uintptr_t)(rowsum + N);
    pb = (pb + 31) & ~(uintptr_t)31;
    unsigned short* pEdgeW1 = (unsigned short*)pb;          // L * 8192
    unsigned short* pEmbW1  = pEdgeW1 + (long)L * 8192;     // L * 4096
    unsigned short* pEmbW2  = pEmbW1  + (long)L * 4096;     // L * 4096
    float* sout   = (float*)d_out;         // [3, N, 64]

    const int T = 256;
    const unsigned ewGrid = (unsigned)((ND + T - 1) / T);
    const unsigned spGrid = (unsigned)((16 * E + T - 1) / T);
    const unsigned emGrid = (unsigned)((E + 127) / 128);
    const unsigned gtGrid = (unsigned)((N + 127) / 128);

    // Pre-pack all layer weights into WMMA fragment-major bf16 layout.
    for (int i = 0; i < L; ++i) {
        pack_b_kernel<<<2, 256, 0, stream>>>(edge_W1 + (long)i * 8192,
                                             pEdgeW1 + (long)i * 8192, 128, 64);
        pack_b_kernel<<<1, 256, 0, stream>>>(emb_W1 + (long)i * 4096,
                                             pEmbW1 + (long)i * 4096, 64, 64);
        pack_b_kernel<<<1, 256, 0, stream>>>(emb_W2 + (long)i * 4096,
                                             pEmbW2 + (long)i * 4096, 64, 64);
    }

    init_kernel<<<ewGrid, T, 0, stream>>>(emb0, e, sout, ND);

    for (int i = 0; i < L; ++i) {
        // ---- branch 0: plain normalized propagation ----
        zero_kernel<<<ewGrid, T, 0, stream>>>(gnn, ND);
        spmm_kernel<<<spGrid, T, 0, stream>>>(G_values, nullptr, nullptr,
                                              h_idx, t_idx, e, gnn, E, 0);
        update_kernel<<<ewGrid, T, 0, stream>>>(e, gnn, sout, ND);

        // ---- branch 1: learned edge reweighting ----
        zero_kernel<<<(unsigned)((N + T - 1) / T), T, 0, stream>>>(rowsum, N);
        edge_mlp_kernel<<<emGrid, 256, 0, stream>>>(
            e + ND, h_idx, t_idx,
            pEdgeW1 + (long)i * 8192, edge_b1 + (long)i * 64,
            edge_W2 + (long)i * 64, edge_b2 + i,
            edge_gumbel + (long)i * E, wbuf, rowsum, (int)E);
        zero_kernel<<<ewGrid, T, 0, stream>>>(gnn, ND);
        spmm_kernel<<<spGrid, T, 0, stream>>>(nullptr, wbuf, rowsum,
                                              h_idx, t_idx, e + ND, gnn, E, 1);
        update_kernel<<<ewGrid, T, 0, stream>>>(e + ND, gnn, sout + ND, ND);

        // ---- branch 2: learned feature gating ----
        gate_kernel<<<gtGrid, 256, 0, stream>>>(
            e + 2 * ND,
            pEmbW1 + (long)i * 4096, emb_b1 + (long)i * 64,
            pEmbW2 + (long)i * 4096, emb_b2 + (long)i * 64,
            emb_gumbel + (long)i * ND, gated, N);
        zero_kernel<<<ewGrid, T, 0, stream>>>(gnn, ND);
        spmm_kernel<<<spGrid, T, 0, stream>>>(G_values, nullptr, nullptr,
                                              h_idx, t_idx, gated, gnn, E, 0);
        update_kernel<<<ewGrid, T, 0, stream>>>(e + 2 * ND, gnn, sout + 2 * ND, ND);
    }
}
